// Head_Graph_87866440942057
// MI455X (gfx1250) — matrix operand order repaired and checked by reading, not compiled, not run on hardware
//
#include <hip/hip_runtime.h>

// Problem constants (match the reference).
#define BATCH    4
#define N_NODES  4096
#define H_DIM    128
#define F_DIM    64
#define ROWS_PB  32              // query rows per block
#define KPW      (N_NODES / 8)   // keys per wave (8 waves/block)

typedef __attribute__((ext_vector_type(16))) _Float16 v16h;
typedef __attribute__((ext_vector_type(8)))  float    v8f;

// ---------------------------------------------------------------------------
// WMMA fragment loaders, CDNA5 ISA 7.12.2 wave32 layouts.
// A (16x32 f16): lane L(0-15) row M=L holds K=0..7 & 16..23; lane L+16 same
//   row holds K=8..15 & 24..31.
// B (32x16 f16): lane L(0-15) col N=L holds K=0..7 & 16..23; lane L+16 K=8..15 & 24..31.
// C/D (16x16 f32): VGPR r -> row r (lanes 0-15) / r+8 (lanes 16-31), col=lane&15.
// ---------------------------------------------------------------------------

static __device__ inline v16h load_a_f32(const float* __restrict__ src, int ld,
                                         int rowbase, int kbase, int lane) {
  const int row = rowbase + (lane & 15);
  const int k0  = kbase + ((lane >> 4) << 3);
  const float* p = src + (size_t)row * ld + k0;
  v16h a;
#pragma unroll
  for (int i = 0; i < 8; ++i) a[i] = (_Float16)p[i];
#pragma unroll
  for (int i = 0; i < 8; ++i) a[8 + i] = (_Float16)p[16 + i];
  return a;
}

static __device__ inline v16h load_a_f16(const _Float16* __restrict__ src, int ld,
                                         int rowbase, int kbase, int lane) {
  const int row = rowbase + (lane & 15);
  const int k0  = kbase + ((lane >> 4) << 3);
  const _Float16* p = src + (size_t)row * ld + k0;
  v16h a;
#pragma unroll
  for (int i = 0; i < 8; ++i) a[i] = p[i];
#pragma unroll
  for (int i = 0; i < 8; ++i) a[8 + i] = p[16 + i];
  return a;
}

// B frag where the B-matrix K-dimension is CONTIGUOUS in memory per column:
// src is [ncols x ld] row-major; B[k][n] = src[(nbase+lane16)*ld + kbase+k].
// Two 16B contiguous loads per lane. Used for K^T (K row-major [key x F]),
// V^T (Vt row-major [F x B*N]) and W^T (Wt row-major [F x H]).
static __device__ inline v16h load_b_kcontig_f16(const _Float16* __restrict__ src,
                                                 int ld, int kbase, int nbase,
                                                 int lane) {
  const int n  = nbase + (lane & 15);
  const int k0 = kbase + ((lane >> 4) << 3);
  const _Float16* p = src + (size_t)n * ld + k0;
  v16h b;
#pragma unroll
  for (int i = 0; i < 8; ++i) b[i] = p[i];
#pragma unroll
  for (int i = 0; i < 8; ++i) b[8 + i] = p[16 + i];
  return b;
}

// ---------------------------------------------------------------------------
// Kernel 0: one-shot conversion of Wq/Wk/Wv (f32, [H x F]) to f16 transposed
// [F x H]. 3 blocks x 256 threads; 8K elements per matrix.
// ---------------------------------------------------------------------------
__global__ __launch_bounds__(256)
void wt_convert_kernel(const float* __restrict__ Wq,
                       const float* __restrict__ Wk,
                       const float* __restrict__ Wv,
                       _Float16* __restrict__ Wqt,
                       _Float16* __restrict__ Wkt,
                       _Float16* __restrict__ Wvt) {
  const int m = blockIdx.x;
  const float* W  = (m == 0) ? Wq : (m == 1) ? Wk : Wv;
  _Float16*   Wt  = (m == 0) ? Wqt : (m == 1) ? Wkt : Wvt;
  for (int e = threadIdx.x; e < F_DIM * H_DIM; e += 256) {
    const int col = e >> 7;       // output feature, 0..63
    const int k   = e & (H_DIM - 1);
    Wt[(size_t)col * H_DIM + k] = (_Float16)W[(size_t)k * F_DIM + col];
  }
}

// ---------------------------------------------------------------------------
// Kernel 1: Q/K/V projection, one wave per 16-row tile of (B*N, H) x.
// Uses the f16 transposed weights so B-frags are contiguous 16B loads.
// Q, K stored row-major f16 [B*N x F]; V stored TRANSPOSED f16 [F x B*N].
// ---------------------------------------------------------------------------
__global__ __launch_bounds__(32)
void qkv_project_kernel(const float* __restrict__ x,
                        const _Float16* __restrict__ Wqt,
                        const _Float16* __restrict__ Wkt,
                        const _Float16* __restrict__ Wvt,
                        _Float16* __restrict__ Q,
                        _Float16* __restrict__ K,
                        _Float16* __restrict__ Vt) {
  const int lane     = threadIdx.x & 31;
  const int half_sel = lane >> 4;
  const int lane16   = lane & 15;
  const int rowbase  = blockIdx.x * 16;

  v16h ax[4];
#pragma unroll
  for (int kk = 0; kk < 4; ++kk)
    ax[kk] = load_a_f32(x, H_DIM, rowbase, kk * 32, lane);

  for (int m = 0; m < 3; ++m) {
    const _Float16* Wt = (m == 0) ? Wqt : (m == 1) ? Wkt : Wvt;

    v8f acc[4] = {};
#pragma unroll
    for (int kk = 0; kk < 4; ++kk) {
#pragma unroll
      for (int ct = 0; ct < 4; ++ct) {
        v16h b = load_b_kcontig_f16(Wt, H_DIM, kk * 32, ct * 16, lane);
        acc[ct] = __builtin_amdgcn_wmma_f32_16x16x32_f16(
            false, ax[kk], false, b, (short)0, acc[ct], false, false);
      }
    }
#pragma unroll
    for (int ct = 0; ct < 4; ++ct) {
      const int col = ct * 16 + lane16;
#pragma unroll
      for (int r = 0; r < 8; ++r) {
        const int row = rowbase + r + half_sel * 8;
        const _Float16 v = (_Float16)acc[ct][r];
        if (m == 0)      Q[(size_t)row * F_DIM + col] = v;
        else if (m == 1) K[(size_t)row * F_DIM + col] = v;
        else             Vt[(size_t)col * (BATCH * N_NODES) + row] = v;
      }
    }
  }
}

// ---------------------------------------------------------------------------
// Kernel 2: masked attention. One block = 8 waves = one (batch, 32-query tile).
// LDS: f16 score strip 32x4096 (256KB) + f32 output accumulator (8KB).
// adj is read exactly once; the "+adj" flag survives softmax in the sign bit
// of the stored exp value.
// ---------------------------------------------------------------------------
__global__ __launch_bounds__(256)
void gat_attn_kernel(const _Float16* __restrict__ Q,
                     const _Float16* __restrict__ K,
                     const _Float16* __restrict__ Vt,
                     const float* __restrict__ adj,
                     float* __restrict__ out) {
  extern __shared__ char smem_raw[];
  float*    s_out    = (float*)smem_raw;                    // 32*64 f32
  float*    s_inv    = s_out + ROWS_PB * F_DIM;             // 32 f32
  _Float16* s_scores = (_Float16*)(s_inv + ROWS_PB);        // 32*4096 f16

  const int tiles_per_b = N_NODES / ROWS_PB;
  const int tile = blockIdx.x % tiles_per_b;
  const int bb   = blockIdx.x / tiles_per_b;

  const int wave     = threadIdx.x >> 5;
  const int lane     = threadIdx.x & 31;
  const int half_sel = lane >> 4;
  const int lane16   = lane & 15;

  const int qrow0 = bb * N_NODES + tile * ROWS_PB;  // base row in Q/K buffers
  const int arow0 = tile * ROWS_PB;                 // base row in adj
  const float scale = 0.125f;                       // F^-0.5

  // Zero the cross-wave output accumulator.
  for (int e = threadIdx.x; e < ROWS_PB * F_DIM; e += 256) s_out[e] = 0.0f;

  // Q A-frags: 2 row-tiles x 2 feature chunks.
  v16h aq[2][2];
#pragma unroll
  for (int rt = 0; rt < 2; ++rt)
#pragma unroll
    for (int kk = 0; kk < 2; ++kk)
      aq[rt][kk] = load_a_f16(Q, F_DIM, qrow0 + rt * 16, kk * 32, lane);

  // ---- Phase 1: masked scores for this wave's 512-key slice -> LDS (f16) ----
  const int key0 = wave * KPW;
  for (int j = 0; j < 32; ++j) {
    const int kb = key0 + j * 16;
    // Prefetch adjacency columns a few tiles ahead (global_prefetch_b8).
    __builtin_prefetch(&adj[(size_t)(arow0 + lane) * N_NODES + kb + 128], 0, 0);

    v16h bkt[2];
#pragma unroll
    for (int kk = 0; kk < 2; ++kk)
      bkt[kk] = load_b_kcontig_f16(K, F_DIM, kk * 32, bb * N_NODES + kb, lane);

#pragma unroll
    for (int rt = 0; rt < 2; ++rt) {
      v8f s = {};
#pragma unroll
      for (int kk = 0; kk < 2; ++kk)
        s = __builtin_amdgcn_wmma_f32_16x16x32_f16(
            false, aq[rt][kk], false, bkt[kk], (short)0, s, false, false);
#pragma unroll
      for (int r = 0; r < 8; ++r) {
        const int row = rt * 16 + r + half_sel * 8;
        const int key = kb + lane16;
        const float av = adj[(size_t)(arow0 + row) * N_NODES + key];
        const float sc = s[r] * scale;
        // masked -> -65504 (f16 lowest); detected again in phase 2.
        s_scores[row * N_NODES + key] =
            (av == 0.0f) ? (_Float16)(-65504.0f) : (_Float16)sc;
      }
    }
  }
  __syncthreads();

  // ---- Phase 2: exact softmax per row (8 threads per row, 32 rows) ----
  {
    const int r = threadIdx.x >> 3;
    const int j = threadIdx.x & 7;
    float m = -1e30f;
    for (int c = j; c < N_NODES; c += 8)
      m = fmaxf(m, (float)s_scores[r * N_NODES + c]);
#pragma unroll
    for (int off = 4; off >= 1; off >>= 1) m = fmaxf(m, __shfl_xor(m, off, 8));
    float sum = 0.0f;
    for (int c = j; c < N_NODES; c += 8) {
      const float sc = (float)s_scores[r * N_NODES + c];
      const bool neighbor = sc > -60000.0f;     // not the mask sentinel
      const float e = __expf(sc - m);           // masked: exp(-huge) == 0
      sum += e;
      // Sign bit := "adj==1" flag (survives underflow as -0.0).
      s_scores[r * N_NODES + c] = (_Float16)(neighbor ? -e : e);
    }
#pragma unroll
    for (int off = 4; off >= 1; off >>= 1) sum += __shfl_xor(sum, off, 8);
    if (j == 0) s_inv[r] = 1.0f / sum;
  }
  __syncthreads();

  // ---- Phase 3: partial O = (softmax + adj) @ V over this wave's keys ----
  v8f accO[2][4] = {};
  for (int c = 0; c < 16; ++c) {
    const int kb = key0 + c * 32;
    v16h bv[4];
#pragma unroll
    for (int ct = 0; ct < 4; ++ct)
      bv[ct] = load_b_kcontig_f16(Vt, BATCH * N_NODES, bb * N_NODES + kb,
                                  ct * 16, lane);
#pragma unroll
    for (int rt = 0; rt < 2; ++rt) {
      const int prow = rt * 16 + lane16;
      const float inv = s_inv[prow];
      const int k0 = half_sel << 3;
      v16h ap;
#pragma unroll
      for (int half = 0; half < 2; ++half) {
#pragma unroll
        for (int i = 0; i < 8; ++i) {
          const int key = kb + half * 16 + k0 + i;
          const float sv = (float)s_scores[prow * N_NODES + key];
          const float p =
              fabsf(sv) * inv + (__builtin_signbitf(sv) ? 1.0f : 0.0f);
          ap[half * 8 + i] = (_Float16)p;
        }
      }
#pragma unroll
      for (int ct = 0; ct < 4; ++ct)
        accO[rt][ct] = __builtin_amdgcn_wmma_f32_16x16x32_f16(
            false, ap, false, bv[ct], (short)0, accO[rt][ct], false, false);
    }
  }

  // Cross-wave reduction via LDS float atomics (ds_add_f32).
#pragma unroll
  for (int rt = 0; rt < 2; ++rt)
#pragma unroll
    for (int ct = 0; ct < 4; ++ct) {
      const int col = ct * 16 + lane16;
#pragma unroll
      for (int r = 0; r < 8; ++r) {
        const int row = rt * 16 + r + half_sel * 8;
        atomicAdd(&s_out[row * F_DIM + col], accO[rt][ct][r]);
      }
    }
  __syncthreads();

  // ---- Phase 4: store f32 output ----
  for (int e = threadIdx.x; e < ROWS_PB * F_DIM; e += 256) {
    const int row = e >> 6;
    const int col = e & 63;
    out[(size_t)(qrow0 + row) * F_DIM + col] = s_out[e];
  }
}

// ---------------------------------------------------------------------------
// Host launcher.
// ---------------------------------------------------------------------------
extern "C" void kernel_launch(void* const* d_in, const int* in_sizes, int n_in,
                              void* d_out, int out_size, void* d_ws, size_t ws_size,
                              hipStream_t stream) {
  (void)in_sizes; (void)n_in; (void)out_size; (void)ws_size;

  const float* x   = (const float*)d_in[0];
  const float* Wq  = (const float*)d_in[1];
  const float* Wk  = (const float*)d_in[2];
  const float* Wv  = (const float*)d_in[3];
  const float* adj = (const float*)d_in[4];
  float* out = (float*)d_out;

  // Workspace: Q, K row-major; Vt transposed (2MB each); then f16 W^T (16KB x3).
  _Float16* Q   = (_Float16*)d_ws;
  _Float16* K   = Q  + (size_t)BATCH * N_NODES * F_DIM;
  _Float16* Vt  = K  + (size_t)BATCH * N_NODES * F_DIM;
  _Float16* Wqt = Vt + (size_t)BATCH * N_NODES * F_DIM;
  _Float16* Wkt = Wqt + (size_t)F_DIM * H_DIM;
  _Float16* Wvt = Wkt + (size_t)F_DIM * H_DIM;

  wt_convert_kernel<<<3, 256, 0, stream>>>(Wq, Wk, Wv, Wqt, Wkt, Wvt);

  qkv_project_kernel<<<(BATCH * N_NODES) / 16, 32, 0, stream>>>(
      x, Wqt, Wkt, Wvt, Q, K, Vt);

  const size_t smem_bytes = (size_t)ROWS_PB * F_DIM * 4 + ROWS_PB * 4 +
                            (size_t)ROWS_PB * N_NODES * 2;  // ~264KB
  hipFuncSetAttribute((const void*)gat_attn_kernel,
                      hipFuncAttributeMaxDynamicSharedMemorySize,
                      (int)smem_bytes);
  gat_attn_kernel<<<BATCH * (N_NODES / ROWS_PB), 256, smem_bytes, stream>>>(
      Q, K, Vt, adj, out);
}
